// JointGNN_81973745811781
// MI455X (gfx1250) — compile-verified, matrix operand order checked by reading, not run on hardware
//
#include <hip/hip_runtime.h>

typedef _Float16 v16h __attribute__((ext_vector_type(16)));
typedef _Float16 v8h  __attribute__((ext_vector_type(8)));
typedef _Float16 v4h  __attribute__((ext_vector_type(4)));
typedef float    v8f  __attribute__((ext_vector_type(8)));

#define TM 32
#define ZSTRIDE 264   // f16 row stride (256 + 8): 528B rows -> 4-bank skew, conflict-free b128 reads
#define HSTRIDE 260   // f32 row stride for h tile

// ---- Prep: split lin1_W (f32, [k][n]) into transposed f16 hi/lo: Wh/Wl[n][k] ----
__global__ __launch_bounds__(256)
void split_transpose_w1(const float* __restrict__ W1,
                        _Float16* __restrict__ Wh, _Float16* __restrict__ Wl) {
    const int k = blockIdx.x;    // 256
    const int n = threadIdx.x;   // 256
    const float w = W1[(size_t)k * 256 + n];
    const _Float16 h = (_Float16)w;
    Wh[(size_t)n * 256 + k] = h;
    Wl[(size_t)n * 256 + k] = (_Float16)(w - (float)h);
}

__device__ __forceinline__ v16h frag16(const _Float16* p) {
    // elements 0..7 = 8 consecutive K at p, elements 8..15 = 8 consecutive K at p+16
    const v8h lo = *(const v8h*)(p);
    const v8h hi = *(const v8h*)(p + 16);
    v16h f;
#pragma unroll
    for (int i = 0; i < 8; ++i) { f[i] = lo[i]; f[i + 8] = hi[i]; }
    return f;
}

__global__ __launch_bounds__(256)
void fused_pair_mlp_wmma(const float* __restrict__ x,        // [N_NODES, 256]
                         const int*   __restrict__ samples,  // [S, 2]
                         const _Float16* __restrict__ Wh,    // [256n][256k] hi split
                         const _Float16* __restrict__ Wl,    // [256n][256k] lo split
                         const float* __restrict__ B1,       // lin1_b [256]
                         const float* __restrict__ W2,       // lin2_W [256,2]
                         const float* __restrict__ B2,       // lin2_b [2]
                         float* __restrict__ out,            // [S, 2]
                         int S)
{
    __shared__ __attribute__((aligned(16))) _Float16 zh[TM * ZSTRIDE];
    __shared__ __attribute__((aligned(16))) _Float16 zl[TM * ZSTRIDE];
    __shared__ __attribute__((aligned(16))) float    hs[TM * HSTRIDE];

    const int tid  = threadIdx.x;
    const int wave = tid >> 5;
    const int lane = tid & 31;
    const int half = lane >> 4;   // lane half selects K sub-range of 16-bit fragments
    const int l16  = lane & 15;
    const int s0   = blockIdx.x * TM;

    // ---------- Stage A: z = x[i0]*x[i1], split into f16 hi/lo tiles in LDS ----------
    {
        const int row  = tid >> 3;                 // 8 threads per row
        const int srow = min(s0 + row, S - 1);
        const int i0 = samples[srow * 2 + 0];
        const int i1 = samples[srow * 2 + 1];
        const float4* xa = (const float4*)(x + (size_t)i0 * 256);
        const float4* xb = (const float4*)(x + (size_t)i1 * 256);
        const int c4base = (tid & 7) * 8;
        _Float16* zhr = zh + row * ZSTRIDE;
        _Float16* zlr = zl + row * ZSTRIDE;
#pragma unroll
        for (int j = 0; j < 8; ++j) {
            const int c4 = c4base + j;
            const float4 a = xa[c4];
            const float4 b = xb[c4];
            const float p0 = a.x * b.x, p1 = a.y * b.y, p2 = a.z * b.z, p3 = a.w * b.w;
            v4h ph, pl;
            ph[0] = (_Float16)p0; pl[0] = (_Float16)(p0 - (float)ph[0]);
            ph[1] = (_Float16)p1; pl[1] = (_Float16)(p1 - (float)ph[1]);
            ph[2] = (_Float16)p2; pl[2] = (_Float16)(p2 - (float)ph[2]);
            ph[3] = (_Float16)p3; pl[3] = (_Float16)(p3 - (float)ph[3]);
            *(v4h*)(zhr + c4 * 4) = ph;
            *(v4h*)(zlr + c4 * 4) = pl;
        }
    }
    __syncthreads();

    // ---------- Stage B: h = relu(z @ W1 + b1) via split-f16 WMMA ----------
    // wave w owns columns [w*32, w*32+32): 2 N-tiles x 2 M-tiles
    const int n0 = wave * 32;
    v8f acc[2][2] = {};

    for (int kc = 0; kc < 256; kc += 32) {
        const int kb = kc + 8 * half;

        // B fragments (32x16 tiles of W1 hi/lo), column-contiguous thanks to transpose
        v16h bh[2], bl[2];
#pragma unroll
        for (int c = 0; c < 2; ++c) {
            const size_t nb = (size_t)(n0 + c * 16 + l16) * 256 + kb;
            bh[c] = frag16(Wh + nb);
            bl[c] = frag16(Wl + nb);
        }
#pragma unroll
        for (int r = 0; r < 2; ++r) {
            const int mb = (r * 16 + l16) * ZSTRIDE + kb;
            const v16h ah = frag16(zh + mb);
            const v16h al = frag16(zl + mb);
#pragma unroll
            for (int c = 0; c < 2; ++c) {
                // a*w ~= ah*bh + ah*bl + al*bh  (lo*lo term ~2^-22, dropped)
                acc[r][c] = __builtin_amdgcn_wmma_f32_16x16x32_f16(
                    false, ah, false, bh[c], (short)0, acc[r][c], false, false);
                acc[r][c] = __builtin_amdgcn_wmma_f32_16x16x32_f16(
                    false, ah, false, bl[c], (short)0, acc[r][c], false, false);
                acc[r][c] = __builtin_amdgcn_wmma_f32_16x16x32_f16(
                    false, al, false, bh[c], (short)0, acc[r][c], false, false);
            }
        }
    }

    // Epilogue: +bias, relu, store h tile to LDS (C/D layout: M = v + 8*half, N = l16)
#pragma unroll
    for (int r = 0; r < 2; ++r) {
#pragma unroll
        for (int c = 0; c < 2; ++c) {
            const int N = n0 + c * 16 + l16;
            const float bias = B1[N];
#pragma unroll
            for (int v = 0; v < 8; ++v) {
                const int M = r * 16 + v + 8 * half;
                hs[M * HSTRIDE + N] = fmaxf(acc[r][c][v] + bias, 0.0f);
            }
        }
    }
    __syncthreads();

    // ---------- Stage C: out = h @ W2 + b2 (2 columns) ----------
    {
        const int row  = tid >> 3;
        const int j    = (tid >> 2) & 1;
        const int part = tid & 3;
        const float* hr = hs + row * HSTRIDE + part * 64;
        const float* wp = W2 + (size_t)part * 64 * 2 + j;
        float s = 0.0f;
#pragma unroll 8
        for (int k = 0; k < 64; ++k)
            s += hr[k] * wp[k * 2];
        s += __shfl_down(s, 1, 32);   // part groups are 4-aligned lanes in a wave32
        s += __shfl_down(s, 2, 32);
        if (part == 0 && (s0 + row) < S)
            out[(size_t)(s0 + row) * 2 + j] = s + B2[j];
    }
}

extern "C" void kernel_launch(void* const* d_in, const int* in_sizes, int n_in,
                              void* d_out, int out_size, void* d_ws, size_t ws_size,
                              hipStream_t stream) {
    // setup_inputs order:
    // 0:x_feature 1:edge_index 2:samples 3:W1 4:b1 5:W2 6:b2 7:lin1_W 8:lin1_b 9:lin2_W 10:lin2_b
    // GNN stack (d_in[1],[3..6]) is dead code in the reference: its output is discarded.
    const float* x       = (const float*)d_in[0];
    const int*   samples = (const int*)  d_in[2];
    const float* lin1_W  = (const float*)d_in[7];
    const float* lin1_b  = (const float*)d_in[8];
    const float* lin2_W  = (const float*)d_in[9];
    const float* lin2_b  = (const float*)d_in[10];
    float* out = (float*)d_out;

    _Float16* Wh = (_Float16*)d_ws;            // 256*256*2 = 128 KB
    _Float16* Wl = Wh + 256 * 256;             // 128 KB   (ws usage: 256 KB)

    const int S = in_sizes[2] / 2; // 100000

    split_transpose_w1<<<256, 256, 0, stream>>>(lin1_W, Wh, Wl);

    const int grid = (S + TM - 1) / TM;
    fused_pair_mlp_wmma<<<grid, 256, 0, stream>>>(x, samples, Wh, Wl,
                                                  lin1_b, lin2_W, lin2_b, out, S);
}
